// EncoderBlock_63625645523181
// MI455X (gfx1250) — compile-verified, matrix operand order checked by reading, not run on hardware
//
#include <hip/hip_runtime.h>
#include <hip/hip_bf16.h>
#include <math.h>

typedef _Float16 v16h __attribute__((ext_vector_type(16)));
typedef _Float16 v8h  __attribute__((ext_vector_type(8)));
typedef float    v8f  __attribute__((ext_vector_type(8)));

#define D_MODEL 768
#define HEADS   12
#define HEAD_DIM 64
#define D_FF    3072
#define SEQ     1024
#define BATCH   8
#define ROWS    (BATCH*SEQ)   // 8192

union FragH { v16h v; v8h h[2]; };

// Async copy global -> LDS (16B per lane), ASYNCcnt-tracked (CDNA5 §08).
// LDS offset = low 32 bits of the flat shared pointer (aperture: addr[31:0]).
__device__ __forceinline__ void async_ld_b128(unsigned lds_off, const void* gaddr) {
  asm volatile("global_load_async_to_lds_b128 %0, %1, off"
               :: "v"(lds_off), "v"((unsigned long long)(size_t)gaddr)
               : "memory");
}
__device__ __forceinline__ void wait_async0() {
  asm volatile("s_wait_asynccnt 0" ::: "memory");
}

// ---------------------------------------------------------------------------
// LayerNorm: one wave (32 lanes) per 768-wide row; unbiased std (ddof=1).
// ---------------------------------------------------------------------------
__global__ void __launch_bounds__(256)
ln_kernel(const float* __restrict__ x, const float* __restrict__ g,
          const float* __restrict__ b, _Float16* __restrict__ out) {
  int wave = threadIdx.x >> 5;
  int lane = threadIdx.x & 31;
  int row  = blockIdx.x * 8 + wave;
  const float* xr = x + (size_t)row * D_MODEL;
  float vals[24];
  float s = 0.f, s2 = 0.f;
#pragma unroll
  for (int i = 0; i < 24; ++i) {
    float vv = xr[lane + i * 32];
    vals[i] = vv; s += vv; s2 += vv * vv;
  }
#pragma unroll
  for (int m = 16; m >= 1; m >>= 1) {
    s  += __shfl_xor(s,  m, 32);
    s2 += __shfl_xor(s2, m, 32);
  }
  float mean = s * (1.0f / 768.0f);
  float var  = (s2 - 768.0f * mean * mean) * (1.0f / 767.0f);
  float stdd = sqrtf(fmaxf(var, 0.0f));
  float inv  = 1.0f / (stdd + 1e-5f);
  float gg = g[0], bb = b[0];
  _Float16* orow = out + (size_t)row * D_MODEL;
#pragma unroll
  for (int i = 0; i < 24; ++i)
    orow[lane + i * 32] = (_Float16)(gg * (vals[i] - mean) * inv + bb);
}

// ---------------------------------------------------------------------------
// Weight convert + transpose: w[K][N] fp32 -> wT[N][K] f16 (so GEMM B loads
// are contiguous along K, matching the 16-bit B-fragment VGPR striping).
// ---------------------------------------------------------------------------
__global__ void __launch_bounds__(256)
wtrans_kernel(const float* __restrict__ w, _Float16* __restrict__ wT,
              int K, int N) {
  long idx = (long)blockIdx.x * 256 + threadIdx.x;
  if (idx >= (long)K * N) return;
  int k = (int)(idx / N), n = (int)(idx % N);
  wT[(size_t)n * K + k] = (_Float16)w[idx];
}

// ---------------------------------------------------------------------------
// WMMA GEMM: C[M,N] = A[M,K](f16) * BT[N,K](f16)^T + bias, fused epilogue.
// EPI 0: -> f16   EPI 1: exact GELU -> f16   EPI 2: + residual -> f32
// Block 256 thr = 8 waves (4 along M x 2 along N); tile 128x128x32.
// Double-buffered LDS fed by global_load_async_to_lds_b128 (ASYNCcnt).
// ---------------------------------------------------------------------------
template <int EPI>
__global__ void __launch_bounds__(256)
gemm_wmma_kernel(const _Float16* __restrict__ A, const _Float16* __restrict__ BT,
                 const float* __restrict__ bias, const float* __restrict__ res,
                 _Float16* __restrict__ outH, float* __restrict__ outF,
                 int M, int N, int K) {
  constexpr int BM = 128, BN = 128, BK = 32, LDT = BK + 8;  // 40 halves/row
  __shared__ _Float16 As[2 * BM * LDT];
  __shared__ _Float16 Bs[2 * BN * LDT];

  const int tid  = threadIdx.x;
  const int lane = tid & 31, wid = tid >> 5;
  const int wm = wid & 3, wn = wid >> 2;           // 4 x 2 wave grid
  const int bRow = blockIdx.y * BM;
  const int bCol = blockIdx.x * BN;

  const v8f vzero = {0.f, 0.f, 0.f, 0.f, 0.f, 0.f, 0.f, 0.f};
  v8f acc[2][4];
#pragma unroll
  for (int i = 0; i < 2; ++i)
#pragma unroll
    for (int j = 0; j < 4; ++j) acc[i][j] = vzero;

  const int lrow = tid >> 2;        // 0..63
  const int lcol = (tid & 3) * 8;   // 0,8,16,24
  const int c0 = (lane < 16) ? 0 : 8;
  const int fr = lane & 15;

  const unsigned laBase = (unsigned)(size_t)&As[lrow * LDT + lcol];
  const unsigned lbBase = (unsigned)(size_t)&Bs[lrow * LDT + lcol];
  auto issue = [&](int k0, int p) {
    unsigned la = laBase + (unsigned)(p * BM * LDT * 2);
    unsigned lb = lbBase + (unsigned)(p * BN * LDT * 2);
    async_ld_b128(la,                A  + (size_t)(bRow + lrow)      * K + k0 + lcol);
    async_ld_b128(la + 64 * LDT * 2, A  + (size_t)(bRow + lrow + 64) * K + k0 + lcol);
    async_ld_b128(lb,                BT + (size_t)(bCol + lrow)      * K + k0 + lcol);
    async_ld_b128(lb + 64 * LDT * 2, BT + (size_t)(bCol + lrow + 64) * K + k0 + lcol);
  };

  const int NIT = K / BK;
  issue(0, 0);
  for (int it = 0; it < NIT; ++it) {
    wait_async0();       // this wave's pending tile is in LDS
    __syncthreads();     // everyone's tile in; previous buffer free
    const int p = it & 1;
    if (it + 1 < NIT) issue((it + 1) * BK, 1 - p);   // overlap with compute

    const _Float16* Ab = &As[p * BM * LDT];
    const _Float16* Bb = &Bs[p * BN * LDT];
    FragH a[2], b[4];
#pragma unroll
    for (int mt = 0; mt < 2; ++mt) {
      int r = wm * 32 + mt * 16 + fr;
      a[mt].h[0] = *(const v8h*)&Ab[r * LDT + c0];
      a[mt].h[1] = *(const v8h*)&Ab[r * LDT + c0 + 16];
    }
#pragma unroll
    for (int nt = 0; nt < 4; ++nt) {
      int c = wn * 64 + nt * 16 + fr;
      b[nt].h[0] = *(const v8h*)&Bb[c * LDT + c0];
      b[nt].h[1] = *(const v8h*)&Bb[c * LDT + c0 + 16];
    }
#pragma unroll
    for (int mt = 0; mt < 2; ++mt)
#pragma unroll
      for (int nt = 0; nt < 4; ++nt)
        acc[mt][nt] = __builtin_amdgcn_wmma_f32_16x16x32_f16(
            false, a[mt].v, false, b[nt].v, (short)0, acc[mt][nt], false, false);
  }

  // Epilogue: C layout -> row = base + r + (lane>=16?8:0), col = base + lane%16
  const int rbase = bRow + wm * 32 + ((lane >= 16) ? 8 : 0);
  const int cb    = bCol + wn * 64 + fr;
#pragma unroll
  for (int mt = 0; mt < 2; ++mt) {
#pragma unroll
    for (int nt = 0; nt < 4; ++nt) {
      int   col = cb + nt * 16;
      float bv  = bias[col];
#pragma unroll
      for (int r = 0; r < 8; ++r) {
        int    row = rbase + mt * 16 + r;
        float  v   = acc[mt][nt][r] + bv;
        size_t off = (size_t)row * N + col;
        if constexpr (EPI == 0) {
          outH[off] = (_Float16)v;
        } else if constexpr (EPI == 1) {
          outH[off] = (_Float16)(0.5f * v * (1.0f + erff(v * 0.70710678118f)));
        } else {
          outF[off] = res[off] + v;
        }
      }
    }
  }
}

// ---------------------------------------------------------------------------
// Flash attention: block = 4 waves, 64 q-rows; KV tiles of 64; online softmax.
// S = Q*K^T and O += P*V both via v_wmma_f32_16x16x32_f16.
// Q and K tiles staged with async global->LDS copies.
// ---------------------------------------------------------------------------
__global__ void __launch_bounds__(128)
flash_kernel(const _Float16* __restrict__ q, const _Float16* __restrict__ k,
             const _Float16* __restrict__ v, _Float16* __restrict__ o) {
  constexpr int D = 64, LDT = 72, TK = 64;
  __shared__ _Float16 qs[64 * LDT];
  __shared__ _Float16 ks[TK * LDT];
  __shared__ _Float16 vts[D * LDT];   // V transposed: [d][kv]
  __shared__ _Float16 ps[64 * LDT];   // P, per-wave 16-row slices

  const int tid = threadIdx.x, lane = tid & 31, wid = tid >> 5;
  const int qt = blockIdx.x, h = blockIdx.y, b = blockIdx.z;
  const size_t rs = D_MODEL;
  const _Float16* qb = q + (size_t)b * SEQ * rs + (size_t)h * D;
  const _Float16* kb = k + (size_t)b * SEQ * rs + (size_t)h * D;
  const _Float16* vb = v + (size_t)b * SEQ * rs + (size_t)h * D;

#pragma unroll
  for (int i = 0; i < 4; ++i) {     // Q tile 64x64 via async copies
    int cidx = tid + i * 128;       // 0..511 b128 chunks
    int r = cidx >> 3, c8 = (cidx & 7) * 8;
    async_ld_b128((unsigned)(size_t)&qs[r * LDT + c8],
                  qb + (size_t)(qt * 64 + r) * rs + c8);
  }
  wait_async0();
  __syncthreads();

  const int c0 = (lane < 16) ? 0 : 8;
  const int fr = lane & 15;
  FragH aq[2];
#pragma unroll
  for (int w2 = 0; w2 < 2; ++w2) {
    int r = wid * 16 + fr;
    aq[w2].h[0] = *(const v8h*)&qs[r * LDT + w2 * 32 + c0];
    aq[w2].h[1] = *(const v8h*)&qs[r * LDT + w2 * 32 + c0 + 16];
  }

  const v8f vzero = {0.f, 0.f, 0.f, 0.f, 0.f, 0.f, 0.f, 0.f};
  float mst[8], lst[8];
  v8f oa[4];
#pragma unroll
  for (int r = 0; r < 8; ++r) { mst[r] = -1e30f; lst[r] = 0.f; }
#pragma unroll
  for (int dt = 0; dt < 4; ++dt) oa[dt] = vzero;

  for (int j = 0; j < SEQ / TK; ++j) {
    __syncthreads();   // all waves done reading previous ks/vts
#pragma unroll
    for (int i = 0; i < 4; ++i) {
      int cidx = tid + i * 128;
      int r = cidx >> 3, c8 = (cidx & 7) * 8;
      async_ld_b128((unsigned)(size_t)&ks[r * LDT + c8],
                    kb + (size_t)(j * TK + r) * rs + c8);
      v8h vv = *(const v8h*)(vb + (size_t)(j * TK + r) * rs + c8);
#pragma unroll
      for (int e = 0; e < 8; ++e) vts[(c8 + e) * LDT + r] = vv[e];
    }
    wait_async0();
    __syncthreads();

    v8f s[4];
#pragma unroll
    for (int nt = 0; nt < 4; ++nt) s[nt] = vzero;
#pragma unroll
    for (int nt = 0; nt < 4; ++nt) {
      int kc = nt * 16 + fr;   // kv row serves as B column
#pragma unroll
      for (int w2 = 0; w2 < 2; ++w2) {
        FragH bf;
        bf.h[0] = *(const v8h*)&ks[kc * LDT + w2 * 32 + c0];
        bf.h[1] = *(const v8h*)&ks[kc * LDT + w2 * 32 + c0 + 16];
        s[nt] = __builtin_amdgcn_wmma_f32_16x16x32_f16(
            false, aq[w2].v, false, bf.v, (short)0, s[nt], false, false);
      }
    }
#pragma unroll
    for (int nt = 0; nt < 4; ++nt)
#pragma unroll
      for (int e = 0; e < 8; ++e) s[nt][e] *= 0.125f;   // 1/sqrt(64)

    // online softmax; xor masks <=8 stay inside each 16-lane column group
#pragma unroll
    for (int r = 0; r < 8; ++r) {
      float rm = fmaxf(fmaxf(s[0][r], s[1][r]), fmaxf(s[2][r], s[3][r]));
#pragma unroll
      for (int m = 8; m >= 1; m >>= 1) rm = fmaxf(rm, __shfl_xor(rm, m, 32));
      float mn = fmaxf(mst[r], rm);
      float corr = expf(mst[r] - mn);
      mst[r] = mn;
      float rsum = 0.f;
#pragma unroll
      for (int nt = 0; nt < 4; ++nt) {
        float p = expf(s[nt][r] - mn);
        s[nt][r] = p;
        rsum += p;
      }
#pragma unroll
      for (int m = 8; m >= 1; m >>= 1) rsum += __shfl_xor(rsum, m, 32);
      lst[r] = lst[r] * corr + rsum;
#pragma unroll
      for (int dt = 0; dt < 4; ++dt) oa[dt][r] *= corr;
      int prow = wid * 16 + r + ((lane >= 16) ? 8 : 0);
#pragma unroll
      for (int nt = 0; nt < 4; ++nt)
        ps[prow * LDT + nt * 16 + fr] = (_Float16)s[nt][r];
    }

    // P (A-layout) from LDS; same-wave LDS is in-order, no barrier needed
    FragH pa[2];
#pragma unroll
    for (int w2 = 0; w2 < 2; ++w2) {
      int r = wid * 16 + fr;
      pa[w2].h[0] = *(const v8h*)&ps[r * LDT + w2 * 32 + c0];
      pa[w2].h[1] = *(const v8h*)&ps[r * LDT + w2 * 32 + c0 + 16];
    }
#pragma unroll
    for (int dt = 0; dt < 4; ++dt) {
      int dc = dt * 16 + fr;
#pragma unroll
      for (int w2 = 0; w2 < 2; ++w2) {
        FragH bv2;
        bv2.h[0] = *(const v8h*)&vts[dc * LDT + w2 * 32 + c0];
        bv2.h[1] = *(const v8h*)&vts[dc * LDT + w2 * 32 + c0 + 16];
        oa[dt] = __builtin_amdgcn_wmma_f32_16x16x32_f16(
            false, pa[w2].v, false, bv2.v, (short)0, oa[dt], false, false);
      }
    }
  }

  _Float16* ob = o + (size_t)b * SEQ * rs + (size_t)h * D;
#pragma unroll
  for (int dt = 0; dt < 4; ++dt)
#pragma unroll
    for (int r = 0; r < 8; ++r) {
      int row = qt * 64 + wid * 16 + r + ((lane >= 16) ? 8 : 0);
      ob[(size_t)row * rs + dt * 16 + fr] = (_Float16)(oa[dt][r] / lst[r]);
    }
}

// ---------------------------------------------------------------------------
extern "C" void kernel_launch(void* const* d_in, const int* in_sizes, int n_in,
                              void* d_out, int out_size, void* d_ws, size_t ws_size,
                              hipStream_t stream) {
  (void)in_sizes; (void)n_in; (void)out_size; (void)ws_size;
  const float* x   = (const float*)d_in[0];
  const float* wq  = (const float*)d_in[1];
  const float* bq  = (const float*)d_in[2];
  const float* wk  = (const float*)d_in[3];
  const float* bk  = (const float*)d_in[4];
  const float* wv  = (const float*)d_in[5];
  const float* bv  = (const float*)d_in[6];
  const float* wo  = (const float*)d_in[7];
  const float* bo  = (const float*)d_in[8];
  const float* w1  = (const float*)d_in[9];
  const float* b1  = (const float*)d_in[10];
  const float* w2  = (const float*)d_in[11];
  const float* b2  = (const float*)d_in[12];
  const float* g1  = (const float*)d_in[13];
  const float* be1 = (const float*)d_in[14];
  const float* g2  = (const float*)d_in[15];
  const float* be2 = (const float*)d_in[16];

  char* ws = (char*)d_ws;
  size_t off = 0;
  auto alloc = [&](size_t bytes) -> void* {
    void* p = ws + off;
    off += (bytes + 255) & ~(size_t)255;
    return p;
  };
  _Float16* ln16   = (_Float16*)alloc((size_t)ROWS * D_MODEL * 2);
  _Float16* q16    = (_Float16*)alloc((size_t)ROWS * D_MODEL * 2);
  _Float16* k16    = (_Float16*)alloc((size_t)ROWS * D_MODEL * 2);
  _Float16* v16    = (_Float16*)alloc((size_t)ROWS * D_MODEL * 2);
  _Float16* attn16 = (_Float16*)alloc((size_t)ROWS * D_MODEL * 2);
  float*    x1     = (float*)   alloc((size_t)ROWS * D_MODEL * 4);
  _Float16* h16    = (_Float16*)alloc((size_t)ROWS * D_FF * 2);
  _Float16* wqT    = (_Float16*)alloc((size_t)D_MODEL * D_MODEL * 2);
  _Float16* wkT    = (_Float16*)alloc((size_t)D_MODEL * D_MODEL * 2);
  _Float16* wvT    = (_Float16*)alloc((size_t)D_MODEL * D_MODEL * 2);
  _Float16* woT    = (_Float16*)alloc((size_t)D_MODEL * D_MODEL * 2);
  _Float16* w1T    = (_Float16*)alloc((size_t)D_MODEL * D_FF * 2);
  _Float16* w2T    = (_Float16*)alloc((size_t)D_FF * D_MODEL * 2);

  long wcnt = (long)D_MODEL * D_MODEL;
  long fcnt = (long)D_MODEL * D_FF;
  wtrans_kernel<<<(wcnt + 255) / 256, 256, 0, stream>>>(wq, wqT, D_MODEL, D_MODEL);
  wtrans_kernel<<<(wcnt + 255) / 256, 256, 0, stream>>>(wk, wkT, D_MODEL, D_MODEL);
  wtrans_kernel<<<(wcnt + 255) / 256, 256, 0, stream>>>(wv, wvT, D_MODEL, D_MODEL);
  wtrans_kernel<<<(wcnt + 255) / 256, 256, 0, stream>>>(wo, woT, D_MODEL, D_MODEL);
  wtrans_kernel<<<(fcnt + 255) / 256, 256, 0, stream>>>(w1, w1T, D_MODEL, D_FF);
  wtrans_kernel<<<(fcnt + 255) / 256, 256, 0, stream>>>(w2, w2T, D_FF, D_MODEL);

  ln_kernel<<<ROWS / 8, 256, 0, stream>>>(x, g1, be1, ln16);

  dim3 g768(D_MODEL / 128, ROWS / 128);
  gemm_wmma_kernel<0><<<g768, 256, 0, stream>>>(ln16, wqT, bq, nullptr, q16, nullptr,
                                                ROWS, D_MODEL, D_MODEL);
  gemm_wmma_kernel<0><<<g768, 256, 0, stream>>>(ln16, wkT, bk, nullptr, k16, nullptr,
                                                ROWS, D_MODEL, D_MODEL);
  gemm_wmma_kernel<0><<<g768, 256, 0, stream>>>(ln16, wvT, bv, nullptr, v16, nullptr,
                                                ROWS, D_MODEL, D_MODEL);

  dim3 gfa(SEQ / 64, HEADS, BATCH);
  flash_kernel<<<gfa, 128, 0, stream>>>(q16, k16, v16, attn16);

  gemm_wmma_kernel<2><<<g768, 256, 0, stream>>>(attn16, woT, bo, x, nullptr, x1,
                                                ROWS, D_MODEL, D_MODEL);

  ln_kernel<<<ROWS / 8, 256, 0, stream>>>(x1, g2, be2, ln16);

  dim3 gff(D_FF / 128, ROWS / 128);
  gemm_wmma_kernel<1><<<gff, 256, 0, stream>>>(ln16, w1T, b1, nullptr, h16, nullptr,
                                               ROWS, D_FF, D_MODEL);
  gemm_wmma_kernel<2><<<g768, 256, 0, stream>>>(h16, w2T, b2, x1, nullptr, (float*)d_out,
                                                ROWS, D_MODEL, D_FF);
}
